// Encoder_9981503996585
// MI455X (gfx1250) — compile-verified, matrix operand order checked by reading, not run on hardware
//
#include <hip/hip_runtime.h>
#include <hip/hip_bf16.h>

// ---- problem constants (match reference) ----
#define D_MODEL   512
#define N_HEADS   8
#define HD        64
#define DIM_FF    2048
#define N_LAYERSC 6
#define BB        4
#define LLEN      2048
#define MROWS     (BB*LLEN)          // 8192
#define KCHUNK    512                // A-tile staging chunk (32 KB LDS for 32 rows)

typedef __attribute__((ext_vector_type(16))) __bf16 v16bf;
typedef __attribute__((ext_vector_type(8)))  float  v8f;
typedef __attribute__((ext_vector_type(4)))  unsigned v4u;
typedef __attribute__((ext_vector_type(8)))  int      v8i;
typedef __attribute__((ext_vector_type(4)))  int      v4i;

#if defined(__has_builtin)
#  if __has_builtin(__builtin_amdgcn_tensor_load_to_lds)
#    define HAVE_TDM 1
#  endif
#endif

#if defined(HAVE_TDM)
#pragma message("CDNA5: TDM path ACTIVE (tensor_load_to_lds builtin available)")
#else
#pragma message("CDNA5: TDM builtin NOT available - using cooperative copy fallback")
#endif

union Frag { unsigned u[8]; v16bf v; };

// K-offset inside a 16-bit 16x32 A/B fragment (ISA 7.12.2):
// VGPR v holds bf16 pair (k0,k0+1), k0 = (v>>2)*16 + half*8 + (v&3)*2
__device__ __forceinline__ int koff(int v, int half) {
  return ((v >> 2) << 4) + (half << 3) + ((v & 3) << 1);
}

__device__ __forceinline__ unsigned short f2bf(float f) {
  union { float f; unsigned u; } x; x.f = f;
  unsigned r = x.u + 0x7FFFu + ((x.u >> 16) & 1u);   // round-to-nearest-even
  return (unsigned short)(r >> 16);
}

// ---------------- weight transpose + bf16 convert: W[l][k][n] -> Wt[l][n][k]
__global__ void k_wt_convert(const float* __restrict__ W, unsigned short* __restrict__ Wt,
                             int K, int N, long total) {
  long i = (long)blockIdx.x * blockDim.x + threadIdx.x;
  if (i >= total) return;
  long per = (long)K * N;
  int  l   = (int)(i / per);
  long rem = i - (long)l * per;
  int  n   = (int)(rem / K);
  int  k   = (int)(rem - (long)n * K);
  Wt[i] = f2bf(W[(long)l * per + (long)k * N + n]);
}

// ---------------- embedding + positional encoding -> x (f32) and xb (bf16)
__global__ void k_embed(const int* __restrict__ src, const float* __restrict__ emb,
                        const float* __restrict__ pe, float* __restrict__ x,
                        unsigned short* __restrict__ xb) {
  long i = (long)blockIdx.x * blockDim.x + threadIdx.x;
  if (i >= (long)MROWS * D_MODEL) return;
  int  d   = (int)(i & (D_MODEL - 1));
  long row = i >> 9;                       // D_MODEL = 512 = 2^9
  int  l   = (int)(row & (LLEN - 1));      // row = b*L + l
  int  tok = src[row];
  float v  = emb[(long)tok * D_MODEL + d] + pe[(long)l * D_MODEL + d];
  x[i]  = v;
  xb[i] = f2bf(v);
}

// ---------------- WMMA GEMM: C[M,N] = A[M,K](bf16) @ Wt[N,K](bf16)^T + bias
// Block = 4 waves sharing one 32-row A tile staged in LDS by the Tensor Data
// Mover (tensor_load_to_lds + s_wait_tensorcnt) in KCHUNK slabs.  Each wave
// computes a 32x64 tile: 2 A fragments (LDS) x 4 B fragments (global, L2-hot)
// -> 8 WMMAs per K-step, so every B fragment feeds two WMMAs.
// mode 0: row-major store (Cf f32 and/or Cbf bf16, optional relu)
// mode 1: "faithful bug" head-permute store -> Qh/Kh [B,H,L,64] bf16
// mode 2: head-permute + transpose store    -> Vt [B,H,64,L] bf16
__global__ void k_gemm(const unsigned short* __restrict__ A,
                       const unsigned short* __restrict__ Wt,
                       const float* __restrict__ bias,
                       float* __restrict__ Cf, unsigned short* __restrict__ Cbf,
                       int M, int N, int K, int mode, int relu) {
  __shared__ unsigned short tileA[32 * KCHUNK];    // 32 KB
  const int lane = threadIdx.x & 31;
  const int wave = threadIdx.x >> 5;
  const int tiles_n = N >> 6;
  const int t0 = blockIdx.x * 4;
  if (t0 >= (M >> 5) * tiles_n) return;            // whole-block guard
  const int tm = t0 / tiles_n;                     // same for all 4 waves
  const int tn = (t0 - tm * tiles_n) + wave;
  const int n = lane & 15, half = lane >> 4;

  v8f z = {};
  v8f acc[2][4];
#pragma unroll
  for (int r = 0; r < 2; ++r)
#pragma unroll
    for (int s = 0; s < 4; ++s) acc[r][s] = z;

  for (int kc = 0; kc < K; kc += KCHUNK) {
    __syncthreads();                               // buffer reuse across slabs
#if defined(HAVE_TDM)
    if (threadIdx.x < 32) {
      // Tensor DMA descriptor (ISA ch.8): 2D tile, 32 rows x KCHUNK cols of 2B data
      const unsigned long long ga =
          (unsigned long long)(A + ((size_t)tm * 32) * K + kc);   // tile start (bytes)
      const unsigned ldsb = (unsigned)(unsigned long long)&tileA[0];
      v4u g0 = { 1u,                                              // count=1, user mode
                 ldsb,                                            // lds_addr
                 (unsigned)(ga & 0xffffffffull),                  // global_addr[31:0]
                 (unsigned)((ga >> 32) & 0x01ffffffull) | 0x80000000u }; // [56:32] | type=2
      v8i g1 = { (int)(1u << 16),                                 // data_size=1 (2 bytes)
                 (int)(((unsigned)K & 0xffffu) << 16),            // tensor_dim0[15:0]
                 (int)((((unsigned)K >> 16) & 0xffffu) | (32u << 16)), // dim0 hi | tensor_dim1=32
                 (int)((unsigned)KCHUNK << 16),                   // tile_dim0 = KCHUNK
                 32,                                              // tile_dim1=32, tile_dim2=0
                 (int)(unsigned)K,                                // tensor_dim0_stride lo
                 0, 0 };
      v4i gz = { 0, 0, 0, 0 };
#if defined(__clang_major__) && __clang_major__ >= 23
      v8i gz8 = { 0, 0, 0, 0, 0, 0, 0, 0 };
      __builtin_amdgcn_tensor_load_to_lds(g0, g1, gz, gz, gz8, 0);
#else
      __builtin_amdgcn_tensor_load_to_lds(g0, g1, gz, gz, 0);
#endif
#if __has_builtin(__builtin_amdgcn_s_wait_tensorcnt)
      __builtin_amdgcn_s_wait_tensorcnt(0);
#else
      asm volatile("s_wait_tensorcnt 0x0" ::: "memory");
#endif
    }
#else
    // fallback: cooperative 128-thread copy, 32 rows x KCHUNK bf16
    for (int idx = threadIdx.x; idx < 32 * (KCHUNK / 8); idx += blockDim.x) {
      const int r = idx >> 6;                      // KCHUNK/8 = 64 uint4 per row
      const int c = idx & 63;
      ((uint4*)tileA)[r * 64 + c] =
          *(const uint4*)(A + ((size_t)(tm * 32 + r)) * K + kc + c * 8);
    }
#endif
    __syncthreads();

    for (int k0 = 0; k0 < KCHUNK; k0 += 32) {
      Frag a0, a1;
#pragma unroll
      for (int v = 0; v < 8; ++v) {
        a0.u[v] = *(const unsigned*)&tileA[n * KCHUNK + k0 + koff(v, half)];          // rows 0..15
        a1.u[v] = *(const unsigned*)&tileA[(n + 16) * KCHUNK + k0 + koff(v, half)];   // rows 16..31
      }
#pragma unroll
      for (int s = 0; s < 4; ++s) {
        const unsigned short* Wrow = Wt + (size_t)(tn * 64 + s * 16 + n) * K + kc + k0;
        Frag b;
#pragma unroll
        for (int v = 0; v < 8; ++v)
          b.u[v] = *(const unsigned*)(Wrow + koff(v, half));
        acc[0][s] = __builtin_amdgcn_wmma_f32_16x16x32_bf16(
            false, a0.v, false, b.v, (short)0, acc[0][s], false, false);
        acc[1][s] = __builtin_amdgcn_wmma_f32_16x16x32_bf16(
            false, a1.v, false, b.v, (short)0, acc[1][s], false, false);
      }
    }
  }

#pragma unroll
  for (int r = 0; r < 2; ++r) {
#pragma unroll
    for (int s = 0; s < 4; ++s) {
      const int col = tn * 64 + s * 16 + n;
      const float bv = bias ? bias[col] : 0.0f;
#pragma unroll
      for (int v = 0; v < 8; ++v) {
        const int row = tm * 32 + r * 16 + v + (half << 3);   // C layout: M = v + 8*half
        float val = acc[r][s][v] + bv;
        if (relu) val = fmaxf(val, 0.0f);
        if (mode == 0) {
          size_t idx = (size_t)row * N + col;
          if (Cf)  Cf[idx]  = val;
          if (Cbf) Cbf[idx] = f2bf(val);
        } else {
          // reshape(B,H,L,hd) with NO transpose: flat f = l*D + col within batch
          const int b_ = row >> 11;            // L = 2048
          const int l  = row & (LLEN - 1);
          const long f = (long)l * D_MODEL + col;
          const int h  = (int)(f >> 17);       // / (L*hd) = 131072
          const int r2 = (int)(f & 131071);
          const int l2 = r2 >> 6;
          const int e  = r2 & 63;
          if (mode == 1)
            Cbf[(((size_t)b_ * N_HEADS + h) * LLEN + l2) * HD + e] = f2bf(val);
          else
            Cbf[(((size_t)b_ * N_HEADS + h) * HD + e) * LLEN + l2] = f2bf(val);
        }
      }
    }
  }
}

// ---------------- streaming (flash) attention, one wave = one 16-row q-tile
// Qh/Kh: [B,H,L,64] bf16 ; Vt: [B,H,64,L] bf16 ; Ob: [B,L,512] bf16 (head-concat)
__global__ void k_flash(const unsigned short* __restrict__ Qh,
                        const unsigned short* __restrict__ Kh,
                        const unsigned short* __restrict__ Vt,
                        unsigned short* __restrict__ Ob) {
  __shared__ unsigned short plds[4][16 * 32];
  const int lane = threadIdx.x & 31;
  const int wave = threadIdx.x >> 5;
  const int w = blockIdx.x * 4 + wave;               // total = B*H*(L/16) = 4096
  const int b_ = w >> 10;
  const int h  = (w >> 7) & 7;
  const int qt = w & 127;
  const int n = lane & 15, half = lane >> 4;
  unsigned short* pl = &plds[wave][0];

  const size_t head = (size_t)b_ * N_HEADS + h;
  const unsigned short* Qrow = Qh + (head * LLEN + qt * 16 + n) * HD;  // m = lane&15
  Frag qa0, qa1;
#pragma unroll
  for (int v = 0; v < 8; ++v) {
    qa0.u[v] = *(const unsigned*)(Qrow + koff(v, half));
    qa1.u[v] = *(const unsigned*)(Qrow + 32 + koff(v, half));
  }

  v8f zz = {};
  v8f o[4];
#pragma unroll
  for (int s = 0; s < 4; ++s) o[s] = zz;
  float mi[8], li[8];
#pragma unroll
  for (int v = 0; v < 8; ++v) { mi[v] = -1.0e30f; li[v] = 0.0f; }

  const unsigned short* Kbase = Kh + head * LLEN * HD;
  const unsigned short* Vbase = Vt + head * HD * LLEN;

  for (int kt = 0; kt < LLEN / 32; ++kt) {
    const int kr = kt * 32;
    // scores: S0 = Q @ K[kr..kr+15]^T, S1 = Q @ K[kr+16..kr+31]^T  (K-dim 64 = 2x32)
    Frag kb;
    const unsigned short* Kr0 = Kbase + (size_t)(kr + n) * HD;
    const unsigned short* Kr1 = Kbase + (size_t)(kr + 16 + n) * HD;
    v8f s0 = zz, s1 = zz;
#pragma unroll
    for (int v = 0; v < 8; ++v) kb.u[v] = *(const unsigned*)(Kr0 + koff(v, half));
    s0 = __builtin_amdgcn_wmma_f32_16x16x32_bf16(false, qa0.v, false, kb.v, (short)0, s0, false, false);
#pragma unroll
    for (int v = 0; v < 8; ++v) kb.u[v] = *(const unsigned*)(Kr0 + 32 + koff(v, half));
    s0 = __builtin_amdgcn_wmma_f32_16x16x32_bf16(false, qa1.v, false, kb.v, (short)0, s0, false, false);
#pragma unroll
    for (int v = 0; v < 8; ++v) kb.u[v] = *(const unsigned*)(Kr1 + koff(v, half));
    s1 = __builtin_amdgcn_wmma_f32_16x16x32_bf16(false, qa0.v, false, kb.v, (short)0, s1, false, false);
#pragma unroll
    for (int v = 0; v < 8; ++v) kb.u[v] = *(const unsigned*)(Kr1 + 32 + koff(v, half));
    s1 = __builtin_amdgcn_wmma_f32_16x16x32_bf16(false, qa1.v, false, kb.v, (short)0, s1, false, false);

    // online softmax; row m = v + 8*half lives on 16 lanes (xor 1/2/4/8 stays in half)
#pragma unroll
    for (int v = 0; v < 8; ++v) {
      float a0 = s0[v] * 0.125f;           // 1/sqrt(64)
      float a1 = s1[v] * 0.125f;
      float mx = fmaxf(a0, a1);
      mx = fmaxf(mx, __shfl_xor(mx, 1, 32));
      mx = fmaxf(mx, __shfl_xor(mx, 2, 32));
      mx = fmaxf(mx, __shfl_xor(mx, 4, 32));
      mx = fmaxf(mx, __shfl_xor(mx, 8, 32));
      const float mn = fmaxf(mi[v], mx);
      const float al = __expf(mi[v] - mn);
      const float p0 = __expf(a0 - mn);
      const float p1 = __expf(a1 - mn);
      float rs = p0 + p1;
      rs += __shfl_xor(rs, 1, 32);
      rs += __shfl_xor(rs, 2, 32);
      rs += __shfl_xor(rs, 4, 32);
      rs += __shfl_xor(rs, 8, 32);
      li[v] = li[v] * al + rs;
      mi[v] = mn;
      o[0][v] *= al; o[1][v] *= al; o[2][v] *= al; o[3][v] *= al;
      const int mrow = v + (half << 3);
      pl[mrow * 32 + n]      = f2bf(p0);   // re-pack P: C-layout -> A-fragment
      pl[mrow * 32 + n + 16] = f2bf(p1);
    }
    asm volatile("s_wait_dscnt 0" ::: "memory");

    Frag pa;
#pragma unroll
    for (int v = 0; v < 8; ++v)
      pa.u[v] = *(const unsigned*)&pl[n * 32 + koff(v, half)];   // m = lane&15

    // O += P(16x32) @ V(32x64); V columns are contiguous in Vt rows
#pragma unroll
    for (int s = 0; s < 4; ++s) {
      Frag vb;
      const unsigned short* Vr = Vbase + (size_t)(s * 16 + n) * LLEN + kr;
#pragma unroll
      for (int v = 0; v < 8; ++v) vb.u[v] = *(const unsigned*)(Vr + koff(v, half));
      o[s] = __builtin_amdgcn_wmma_f32_16x16x32_bf16(
          false, pa.v, false, vb.v, (short)0, o[s], false, false);
    }
  }

  // normalize + transpose(0,2,1,3) concat store: Ob[b, l2, h*64+e]
#pragma unroll
  for (int s = 0; s < 4; ++s) {
#pragma unroll
    for (int v = 0; v < 8; ++v) {
      const float val = o[s][v] / li[v];
      const int l2  = qt * 16 + v + (half << 3);
      const int col = h * HD + s * 16 + n;
      Ob[((size_t)b_ * LLEN + l2) * D_MODEL + col] = f2bf(val);
    }
  }
}

// ---------------- residual + layernorm; writes f32 state and bf16 copy
__global__ void k_resid_ln(const float* __restrict__ x, const float* __restrict__ r,
                           const float* __restrict__ g, const float* __restrict__ be,
                           float* __restrict__ xout, unsigned short* __restrict__ xb) {
  __shared__ float red[256];
  const int row = blockIdx.x, t = threadIdx.x;
  const float* xr = x + (size_t)row * D_MODEL;
  const float* rr = r + (size_t)row * D_MODEL;
  const float a0 = xr[t] + rr[t];
  const float a1 = xr[t + 256] + rr[t + 256];
  red[t] = a0 + a1;
  __syncthreads();
  for (int o2 = 128; o2 > 0; o2 >>= 1) { if (t < o2) red[t] += red[t + o2]; __syncthreads(); }
  const float mu = red[0] * (1.0f / D_MODEL);
  __syncthreads();
  const float d0 = a0 - mu, d1 = a1 - mu;
  red[t] = d0 * d0 + d1 * d1;
  __syncthreads();
  for (int o2 = 128; o2 > 0; o2 >>= 1) { if (t < o2) red[t] += red[t + o2]; __syncthreads(); }
  const float rs = rsqrtf(red[0] * (1.0f / D_MODEL) + 1e-5f);
  const float y0 = d0 * rs * g[t]       + be[t];
  const float y1 = d1 * rs * g[t + 256] + be[t + 256];
  xout[(size_t)row * D_MODEL + t]       = y0;
  xout[(size_t)row * D_MODEL + t + 256] = y1;
  xb[(size_t)row * D_MODEL + t]         = f2bf(y0);
  xb[(size_t)row * D_MODEL + t + 256]   = f2bf(y1);
}

__global__ void k_copy(const float* __restrict__ src, float* __restrict__ dst, long n) {
  long i = (long)blockIdx.x * blockDim.x + threadIdx.x;
  if (i < n) dst[i] = src[i];
}

extern "C" void kernel_launch(void* const* d_in, const int* in_sizes, int n_in,
                              void* d_out, int out_size, void* d_ws, size_t ws_size,
                              hipStream_t stream) {
  (void)in_sizes; (void)n_in; (void)out_size; (void)ws_size;
  const int*   src = (const int*)  d_in[0];
  const float* emb = (const float*)d_in[1];
  const float* pe  = (const float*)d_in[2];
  const float* Wq  = (const float*)d_in[3];  const float* bq  = (const float*)d_in[4];
  const float* Wk  = (const float*)d_in[5];  const float* bk  = (const float*)d_in[6];
  const float* Wv  = (const float*)d_in[7];  const float* bv  = (const float*)d_in[8];
  const float* Wo  = (const float*)d_in[9];  const float* bo  = (const float*)d_in[10];
  const float* W1  = (const float*)d_in[11]; const float* b1  = (const float*)d_in[12];
  const float* W2  = (const float*)d_in[13]; const float* b2  = (const float*)d_in[14];
  const float* g1  = (const float*)d_in[15]; const float* be1 = (const float*)d_in[16];
  const float* g2  = (const float*)d_in[17]; const float* be2 = (const float*)d_in[18];
  float* out = (float*)d_out;

  char* base = (char*)d_ws;
  size_t off = 0;
  auto alloc = [&](size_t bytes) -> void* {
    off = (off + 255) & ~(size_t)255;
    void* p = base + off;
    off += bytes;
    return p;
  };
  const size_t wsz  = (size_t)N_LAYERSC * 512 * 512;
  const size_t wfsz = (size_t)N_LAYERSC * 512 * 2048;
  unsigned short* WqT = (unsigned short*)alloc(wsz * 2);
  unsigned short* WkT = (unsigned short*)alloc(wsz * 2);
  unsigned short* WvT = (unsigned short*)alloc(wsz * 2);
  unsigned short* WoT = (unsigned short*)alloc(wsz * 2);
  unsigned short* W1T = (unsigned short*)alloc(wfsz * 2);
  unsigned short* W2T = (unsigned short*)alloc(wfsz * 2);
  const size_t xsz = (size_t)MROWS * D_MODEL;
  float*          x     = (float*)alloc(xsz * 4);
  unsigned short* xb    = (unsigned short*)alloc(xsz * 2);
  unsigned short* Qh    = (unsigned short*)alloc(xsz * 2);
  unsigned short* Kh    = (unsigned short*)alloc(xsz * 2);
  unsigned short* Vt    = (unsigned short*)alloc(xsz * 2);
  unsigned short* Ob    = (unsigned short*)alloc(xsz * 2);
  float*          attnf = (float*)alloc(xsz * 4);
  unsigned short* ff1b  = (unsigned short*)alloc((size_t)MROWS * DIM_FF * 2);
  float*          ff2f  = (float*)alloc(xsz * 4);

  k_wt_convert<<<(int)((wsz  + 255) / 256), 256, 0, stream>>>(Wq, WqT, 512,  512,  (long)wsz);
  k_wt_convert<<<(int)((wsz  + 255) / 256), 256, 0, stream>>>(Wk, WkT, 512,  512,  (long)wsz);
  k_wt_convert<<<(int)((wsz  + 255) / 256), 256, 0, stream>>>(Wv, WvT, 512,  512,  (long)wsz);
  k_wt_convert<<<(int)((wsz  + 255) / 256), 256, 0, stream>>>(Wo, WoT, 512,  512,  (long)wsz);
  k_wt_convert<<<(int)((wfsz + 255) / 256), 256, 0, stream>>>(W1, W1T, 512,  2048, (long)wfsz);
  k_wt_convert<<<(int)((wfsz + 255) / 256), 256, 0, stream>>>(W2, W2T, 2048, 512,  (long)wfsz);

  k_embed<<<(int)((xsz + 255) / 256), 256, 0, stream>>>(src, emb, pe, x, xb);

  const int gD  = (MROWS / 32) * (D_MODEL / 64) / 4;   // 512 blocks of 4 waves
  const int gF1 = (MROWS / 32) * (DIM_FF  / 64) / 4;   // 2048
  const int gAt = (BB * N_HEADS * (LLEN / 16)) / 4;    // 1024

  for (int i = 0; i < N_LAYERSC; ++i) {
    const unsigned short* wqT = WqT + (size_t)i * 512 * 512;
    const unsigned short* wkT = WkT + (size_t)i * 512 * 512;
    const unsigned short* wvT = WvT + (size_t)i * 512 * 512;
    const unsigned short* woT = WoT + (size_t)i * 512 * 512;
    const unsigned short* w1T = W1T + (size_t)i * 512 * 2048;
    const unsigned short* w2T = W2T + (size_t)i * 2048 * 512;

    k_gemm<<<gD, 128, 0, stream>>>(xb, wqT, bq + i * 512, nullptr, Qh, MROWS, 512, 512, 1, 0);
    k_gemm<<<gD, 128, 0, stream>>>(xb, wkT, bk + i * 512, nullptr, Kh, MROWS, 512, 512, 1, 0);
    k_gemm<<<gD, 128, 0, stream>>>(xb, wvT, bv + i * 512, nullptr, Vt, MROWS, 512, 512, 2, 0);
    k_flash<<<gAt, 128, 0, stream>>>(Qh, Kh, Vt, Ob);
    k_gemm<<<gD, 128, 0, stream>>>(Ob, woT, bo + i * 512, attnf, nullptr, MROWS, 512, 512, 0, 0);
    k_resid_ln<<<MROWS, 256, 0, stream>>>(x, attnf, g1 + i * 512, be1 + i * 512, x, xb);
    k_gemm<<<gF1, 128, 0, stream>>>(xb, w1T, b1 + i * DIM_FF, nullptr, ff1b, MROWS, DIM_FF, 512, 0, 1);
    k_gemm<<<gD, 128, 0, stream>>>(ff1b, w2T, b2 + i * 512, ff2f, nullptr, MROWS, 512, DIM_FF, 0, 0);
    k_resid_ln<<<MROWS, 256, 0, stream>>>(x, ff2f, g2 + i * 512, be2 + i * 512, x, xb);
  }

  k_copy<<<(int)((xsz + 255) / 256), 256, 0, stream>>>(x, out, (long)xsz);
}
